// GraphSequenceModel_69191923138562
// MI455X (gfx1250) — matrix operand order repaired and checked by reading, not compiled, not run on hardware
//
#include <hip/hip_runtime.h>
#include <hip/hip_bf16.h>
#include <math.h>

typedef unsigned int u32;
typedef unsigned short u16;
typedef long long i64;

typedef __attribute__((ext_vector_type(16))) __bf16 v16bf;
typedef __attribute__((ext_vector_type(8)))  float  v8f;

struct __align__(16) U128 { u32 d[4]; };
union FragBF { v16bf v; U128 q[2]; };

#define N_NODES   50000
#define N_EDGES   1600000
#define T_STEPS   8
#define HID       128
#define GDIM      512      // 4 gates * 128
#define NUM_GRAPHS 64
#define OUT_DIM   16
#define LDS_STRIDE 516     // 512 + 4 floats pad: de-conflicts rows 0..7 vs 8..15

__device__ __forceinline__ u16 f2bf(float f) {
  u32 u = __float_as_uint(f);
  u32 r = u + 0x7FFFu + ((u >> 16) & 1u);   // round-to-nearest-even
  return (u16)(r >> 16);
}

// ---------------- generic helpers ----------------
__global__ void gsl_zero_u32(u32* __restrict__ p, i64 n) {
  i64 i = (i64)blockIdx.x * blockDim.x + threadIdx.x;
  if (i < n) p[i] = 0u;
}

__global__ void gsl_cvt_f32_bf16(const float* __restrict__ in, u16* __restrict__ out, i64 n) {
  i64 i = (i64)blockIdx.x * blockDim.x + threadIdx.x;
  if (i < n) out[i] = f2bf(in[i]);
}

// ---------------- graph normalization ----------------
__global__ void gsl_deg_count(const i64* __restrict__ ei, float* __restrict__ deg, i64 E) {
  i64 e = (i64)blockIdx.x * blockDim.x + threadIdx.x;
  if (e >= E) return;
  atomicAdd(&deg[(size_t)ei[e]], 1.0f);
}

__global__ void gsl_dinv(const float* __restrict__ deg, float* __restrict__ dinv, i64 n) {
  i64 i = (i64)blockIdx.x * blockDim.x + threadIdx.x;
  if (i >= n) return;
  float d = deg[i];
  dinv[i] = (d > 0.0f) ? rsqrtf(fmaxf(d, 1.0f)) : 0.0f;
}

__global__ void gsl_edge_norm(const i64* __restrict__ ei, const float* __restrict__ dinv,
                              float* __restrict__ nrm, i64 E) {
  i64 e = (i64)blockIdx.x * blockDim.x + threadIdx.x;
  if (e >= E) return;
  nrm[e] = dinv[(size_t)ei[e]] * dinv[(size_t)ei[E + e]];
}

// ---------------- weight packing into WMMA B-fragment order ----------------
// B tile = 32(K) x 16(Nout), stored lane-major: 32 lanes * 16 bf16 each.
// lane L (0-15):  col N = L,    elems 0-7 hold K 0..7, elems 8-15 hold K 16..23
// lane L (16-31): col N = L-16, elems 0-7 hold K 8..15, elems 8-15 hold K 24..31
// Big K (512): block0=Wx[:,0], block1=Wx[:,1], block2=Wh[:,0], block3=Wh[:,1]
// Big N (512): gate g = N/128, col = N%128.
__global__ void gsl_pack_weights(const float* __restrict__ Wx, const float* __restrict__ Wh,
                                 u16* __restrict__ Wp) {
  int tile = blockIdx.x;            // 0..511 : kb*32 + nb
  int kb = tile >> 5;               // 0..15
  int nb = tile & 31;               // 0..31
  int lane = threadIdx.x;           // 0..31
  int Ncol = nb * 16 + (lane & 15);
  int g = Ncol >> 7;
  int cc = Ncol & 127;
  u16* outp = Wp + (((size_t)tile * 32 + lane) << 4);
  #pragma unroll
  for (int i = 0; i < 16; ++i) {
    int j = i >> 1, e = i & 1;
    int Kl = ((lane >= 16) ? 8 : 0) + (i & 7) + ((i >= 8) ? 16 : 0);
    int kk = kb * 32 + Kl;          // 0..511
    int blk = kk >> 7;              // 0..3
    int r = kk & 127;
    float w;
    if (blk < 2) w = Wx[(((size_t)g * 2 + blk) * 128 + r) * 128 + cc];
    else         w = Wh[(((size_t)g * 2 + (blk - 2)) * 128 + r) * 128 + cc];
    outp[i] = f2bf(w);
  }
}

__global__ void gsl_gate_bias(const float* __restrict__ bx, const float* __restrict__ bh,
                              const float* __restrict__ bg, float* __restrict__ gb) {
  int i = blockIdx.x * blockDim.x + threadIdx.x;
  if (i < GDIM) gb[i] = bx[i] + bh[i] + bg[i];
}

// ---------------- fused dual SpMM: hx = -A_hat*xt ; hh = -A_hat*h ----------------
__global__ void gsl_spmm(const i64* __restrict__ ei, const float* __restrict__ nrm,
                         const float* __restrict__ xt, const float* __restrict__ h,
                         float* __restrict__ hx, float* __restrict__ hh, i64 E) {
  i64 tid = (i64)blockIdx.x * blockDim.x + threadIdx.x;
  i64 e = tid >> 2;
  if (e >= E) return;
  int part = (int)(tid & 3) * 32;
  size_t s = (size_t)ei[e], d = (size_t)ei[E + e];
  float w = -nrm[e];
  const float* xs = xt + s * HID + part;
  const float* hs = h  + s * HID + part;
  float* xd = hx + d * HID + part;
  float* hd = hh + d * HID + part;
  #pragma unroll
  for (int f = 0; f < 32; ++f) {
    atomicAdd(xd + f, w * xs[f]);
    atomicAdd(hd + f, w * hs[f]);
  }
}

// ---------------- fused WMMA GEMM + LSTM cell ----------------
// Block = 16 node rows. 8 waves, each wave owns 64 gate cols (4 16x16 tiles),
// K = 512 (xt | L_hat*xt | h | L_hat*h), fully unrolled: 64 WMMAs/wave.
// Epilogue: accumulators -> LDS (16 x 512 stripe), barrier, peephole LSTM
// pointwise math, writes c / h / bf16(h) for this block's 16 rows only.
__global__ void __launch_bounds__(256)
gsl_gemm_cell(const u16* __restrict__ xt_bf, const u16* __restrict__ hx_bf,
              const u16* __restrict__ h_bf,  const u16* __restrict__ hh_bf,
              const u16* __restrict__ Wp, const float* __restrict__ gb,
              const float* __restrict__ wc, float* __restrict__ c,
              float* __restrict__ h, u16* __restrict__ hbf) {
  __shared__ float As[16 * LDS_STRIDE];

  int lane = threadIdx.x & 31;
  int wave = threadIdx.x >> 5;          // 0..7
  int mrow = blockIdx.x * 16;
  size_t rA = (size_t)(mrow + (lane & 15)) * HID;
  int off0 = (lane & 16) ? 8 : 0;       // A layout: lanes 0-15 K{0..7,16..23}, 16-31 K{8..15,24..31}

  v8f zero8 = {0.f,0.f,0.f,0.f,0.f,0.f,0.f,0.f};
  v8f acc[4];
  #pragma unroll
  for (int t = 0; t < 4; ++t) acc[t] = zero8;

  #pragma unroll
  for (int kb = 0; kb < 16; ++kb) {
    int blk = kb >> 2;
    const u16* Asrc = (blk == 0) ? xt_bf : (blk == 1) ? hx_bf : (blk == 2) ? h_bf : hh_bf;
    int klocal = (kb & 3) << 5;
    // Issue A + all 4 B fragment loads up front so they clause together.
    FragBF a;
    const U128* ap = (const U128*)(Asrc + rA + klocal + off0);
    a.q[0] = ap[0];                     // K base..base+7
    a.q[1] = ap[2];                     // K base+16..base+23 (32B ahead)
    FragBF b[4];
    #pragma unroll
    for (int t = 0; t < 4; ++t) {
      int nb = wave * 4 + t;
      const U128* bp = (const U128*)(Wp + (((size_t)(kb * 32 + nb) * 32 + lane) << 4));
      b[t].q[0] = bp[0];
      b[t].q[1] = bp[1];
    }
    #pragma unroll
    for (int t = 0; t < 4; ++t) {
      acc[t] = __builtin_amdgcn_wmma_f32_16x16x32_bf16(
          false, a.v, false, b[t].v, (short)0, acc[t], false, false);
    }
  }

  // C layout: VGPR r -> lanes 0-15 row r, lanes 16-31 row 8+r; col = lane&15.
  int rowhi = (lane & 16) ? 8 : 0;
  #pragma unroll
  for (int t = 0; t < 4; ++t) {
    int col = (wave * 4 + t) * 16 + (lane & 15);
    #pragma unroll
    for (int r = 0; r < 8; ++r) {
      As[(rowhi + r) * LDS_STRIDE + col] = acc[t][r];
    }
  }
  __syncthreads();

  // Pointwise peephole LSTM: 16 rows x 128 hid = 2048 elems, 8 per thread.
  int row = threadIdx.x >> 4;           // 0..15
  int hd0 = (threadIdx.x & 15) * 8;     // 0,8,...,120
  const float* arow = &As[row * LDS_STRIDE];
  size_t base = (size_t)(mrow + row) * HID + hd0;
  #pragma unroll
  for (int k = 0; k < 8; ++k) {
    int hd = hd0 + k;
    float a0 = arow[hd]       + gb[hd];
    float a1 = arow[128 + hd] + gb[128 + hd];
    float a2 = arow[256 + hd] + gb[256 + hd];
    float a3 = arow[384 + hd] + gb[384 + hd];
    float cv = c[base + k];
    float ig = 1.0f / (1.0f + __expf(-(a0 + wc[hd] * cv)));
    float fg = 1.0f / (1.0f + __expf(-(a1 + wc[128 + hd] * cv)));
    float tg = tanhf(a2);
    float cn = fg * cv + ig * tg;
    float og = 1.0f / (1.0f + __expf(-(a3 + wc[256 + hd] * cn)));
    float hn = og * tanhf(cn);
    c[base + k] = cn;
    h[base + k] = hn;
    hbf[base + k] = f2bf(hn);
  }
}

// ---------------- pooling + classifier ----------------
__global__ void gsl_pool_feat(const float* __restrict__ h, const i64* __restrict__ batch,
                              float* __restrict__ pooled, i64 n) {
  i64 idx = (i64)blockIdx.x * blockDim.x + threadIdx.x;
  if (idx >= n) return;
  size_t node = (size_t)(idx >> 7);
  int hd = (int)(idx & 127);
  int g = (int)batch[node];
  atomicAdd(&pooled[(size_t)g * HID + hd], h[idx]);
}

__global__ void gsl_pool_cnt(const i64* __restrict__ batch, float* __restrict__ cnt, i64 n) {
  i64 i = (i64)blockIdx.x * blockDim.x + threadIdx.x;
  if (i >= n) return;
  atomicAdd(&cnt[(size_t)batch[i]], 1.0f);
}

__global__ void gsl_classify(const float* __restrict__ pooled, const float* __restrict__ cnt,
                             const float* __restrict__ W, const float* __restrict__ b,
                             float* __restrict__ out) {
  int tid = blockIdx.x * blockDim.x + threadIdx.x;
  if (tid >= NUM_GRAPHS * OUT_DIM) return;
  int g = tid >> 4, o = tid & 15;
  float inv = 1.0f / fmaxf(cnt[g], 1.0f);
  float s = 0.0f;
  #pragma unroll 16
  for (int hd = 0; hd < HID; ++hd)
    s += pooled[(size_t)g * HID + hd] * inv * W[hd * OUT_DIM + o];
  out[tid] = s + b[o];
}

extern "C" void kernel_launch(void* const* d_in, const int* in_sizes, int n_in,
                              void* d_out, int out_size, void* d_ws, size_t ws_size,
                              hipStream_t stream) {
  (void)in_sizes; (void)n_in; (void)out_size; (void)ws_size;
  const float* x    = (const float*)d_in[0];
  const i64*   ei   = (const i64*)d_in[1];
  const i64*   batch= (const i64*)d_in[2];
  const float* Wx   = (const float*)d_in[3];
  const float* bx   = (const float*)d_in[4];
  const float* Wh   = (const float*)d_in[5];
  const float* bh   = (const float*)d_in[6];
  const float* wc   = (const float*)d_in[7];
  const float* bg   = (const float*)d_in[8];
  const float* clfW = (const float*)d_in[9];
  const float* clfb = (const float*)d_in[10];
  float* out = (float*)d_out;

  char* cur = (char*)d_ws;
  auto alloc = [&](size_t bytes) -> void* {
    void* p = (void*)cur;
    cur += (bytes + 255) & ~(size_t)255;
    return p;
  };
  float* deg    = (float*)alloc((size_t)N_NODES * 4);
  float* dinv   = (float*)alloc((size_t)N_NODES * 4);
  float* nrm    = (float*)alloc((size_t)N_EDGES * 4);
  float* gb     = (float*)alloc((size_t)GDIM * 4);
  u16*   Wp     = (u16*)  alloc((size_t)GDIM * GDIM * 2);
  float* h      = (float*)alloc((size_t)N_NODES * HID * 4);   // h, c adjacent (256B-mult sizes)
  float* c      = (float*)alloc((size_t)N_NODES * HID * 4);
  float* hx     = (float*)alloc((size_t)N_NODES * HID * 4);   // hx, hh adjacent
  float* hh     = (float*)alloc((size_t)N_NODES * HID * 4);
  u16*   xt_bf  = (u16*)  alloc((size_t)N_NODES * HID * 2);
  u16*   hx_bf  = (u16*)  alloc((size_t)N_NODES * HID * 2);   // hx_bf, hh_bf adjacent
  u16*   hh_bf  = (u16*)  alloc((size_t)N_NODES * HID * 2);
  u16*   h_bf   = (u16*)  alloc((size_t)N_NODES * HID * 2);
  float* pooled = (float*)alloc((size_t)NUM_GRAPHS * HID * 4); // pooled, cnt adjacent
  float* cnt    = (float*)alloc((size_t)NUM_GRAPHS * 4);
  (void)c; (void)hh; (void)hh_bf; (void)cnt;

  const int TPB = 256;
  auto nblk = [](i64 n) { return (int)((n + 255) / 256); };
  const i64 NH = (i64)N_NODES * HID;

  // --- one-time setup (re-done every call for determinism) ---
  gsl_zero_u32<<<nblk(N_NODES), TPB, 0, stream>>>((u32*)deg, N_NODES);
  gsl_deg_count<<<nblk(N_EDGES), TPB, 0, stream>>>(ei, deg, N_EDGES);
  gsl_dinv<<<nblk(N_NODES), TPB, 0, stream>>>(deg, dinv, N_NODES);
  gsl_edge_norm<<<nblk(N_EDGES), TPB, 0, stream>>>(ei, dinv, nrm, N_EDGES);
  gsl_pack_weights<<<512, 32, 0, stream>>>(Wx, Wh, Wp);
  gsl_gate_bias<<<2, TPB, 0, stream>>>(bx, bh, bg, gb);
  gsl_zero_u32<<<nblk(2 * NH), TPB, 0, stream>>>((u32*)h, 2 * NH);         // h and c
  gsl_zero_u32<<<nblk(NH / 2), TPB, 0, stream>>>((u32*)h_bf, NH / 2);      // bf16 h

  // --- time loop ---
  for (int t = 0; t < T_STEPS; ++t) {
    const float* xt = x + (size_t)t * N_NODES * HID;
    gsl_zero_u32<<<nblk(2 * NH), TPB, 0, stream>>>((u32*)hx, 2 * NH);      // hx and hh
    gsl_cvt_f32_bf16<<<nblk(NH), TPB, 0, stream>>>(xt, xt_bf, NH);
    gsl_spmm<<<nblk((i64)4 * N_EDGES), TPB, 0, stream>>>(ei, nrm, xt, h, hx, hh, N_EDGES);
    gsl_cvt_f32_bf16<<<nblk(2 * NH), TPB, 0, stream>>>(hx, hx_bf, 2 * NH); // hx+hh -> bf16
    gsl_gemm_cell<<<N_NODES / 16, 256, 0, stream>>>(xt_bf, hx_bf, h_bf, hh_bf,
                                                    Wp, gb, wc, c, h, h_bf);
  }

  // --- pooling + classifier ---
  gsl_zero_u32<<<nblk(NUM_GRAPHS * HID + NUM_GRAPHS), TPB, 0, stream>>>(
      (u32*)pooled, NUM_GRAPHS * HID + NUM_GRAPHS);
  gsl_pool_feat<<<nblk(NH), TPB, 0, stream>>>(h, batch, pooled, NH);
  gsl_pool_cnt<<<nblk(N_NODES), TPB, 0, stream>>>(batch, cnt, N_NODES);
  gsl_classify<<<4, TPB, 0, stream>>>(pooled, cnt, clfW, clfb, out);
}
